// UnrolledLSTM_1357209666457
// MI455X (gfx1250) — matrix operand order repaired and checked by reading, compiled, not run on hardware
//
#include <hip/hip_runtime.h>

// Problem constants (match reference)
#define BB 32
#define TT 1024
#define II 256
#define HH 256
#define GG 1024          // 4*H
#define KCAT 512         // H + I  (concat [h | x_t])
#define NTILES 64        // GG/16
#define KSTEPS 16        // KCAT/32
#define GSTR 1028        // padded gate-buffer row stride (floats)

typedef __attribute__((ext_vector_type(16))) _Float16 v16h;
typedef __attribute__((ext_vector_type(8)))  float    v8f;

// ---------------------------------------------------------------------------
// Prep kernel: pack Bcat = [W_hh ; W_ih] (per direction) into f16 fragments in
// the exact V_WMMA_F32_16X16X32_F16 B-matrix lane layout, and sum the biases.
//
// wpack layout: [d][nt][kk][lane][16 halves]
//   element (d,nt,kk,L,h) = Bcat_d[k][g],
//   k = kk*32 + (h&8)*2 + (L>>4)*8 + (h&7)   (B layout: lanes 0-15 hold K=0..7,16..23,
//                                   lanes 16-31 hold K=8..15,24..31 of the block)
//   g = nt*16 + (L&15)
//   Bcat_d[k][g] = (k < H) ? w_hh_d[g][k] : w_ih_d[g][k-H]
// ---------------------------------------------------------------------------
__global__ __launch_bounds__(256) void lstm_prep(
    const float* __restrict__ w_ih_f, const float* __restrict__ w_hh_f,
    const float* __restrict__ b_ih_f, const float* __restrict__ b_hh_f,
    const float* __restrict__ w_ih_b, const float* __restrict__ w_hh_b,
    const float* __restrict__ b_ih_b, const float* __restrict__ b_hh_b,
    _Float16* __restrict__ wpack, float* __restrict__ bias)
{
    const int t0 = blockIdx.x * blockDim.x + threadIdx.x;   // 0 .. 262143
    const int nthreads = 1024 * 256;
#pragma unroll
    for (int e = 0; e < 4; ++e) {
        int idx = t0 + e * nthreads;                        // 0 .. 2^20-1
        int h  = idx & 15;
        int L  = (idx >> 4) & 31;
        int kk = (idx >> 9) & 15;
        int nt = (idx >> 13) & 63;
        int d  = (idx >> 19) & 1;
        int k  = kk * 32 + ((h & 8) << 1) + ((L >> 4) << 3) + (h & 7);
        int g  = nt * 16 + (L & 15);
        const float* wih = d ? w_ih_b : w_ih_f;
        const float* whh = d ? w_hh_b : w_hh_f;
        float v = (k < HH) ? whh[g * HH + k] : wih[g * II + (k - HH)];
        wpack[idx] = (_Float16)v;
    }
    if (t0 < 2 * GG) {
        int d = t0 >> 10, g = t0 & (GG - 1);
        bias[t0] = d ? (b_ih_b[g] + b_hh_b[g]) : (b_ih_f[g] + b_hh_f[g]);
    }
}

// ---------------------------------------------------------------------------
// Persistent recurrent kernel. Grid = 4 blocks: (direction, batch-tile).
// 1024 threads = 32 wave32 waves. Each wave owns 2 N-tiles (32 gate columns).
// Per step: gates = [h|x_t](16x512) @ Bcat(512x1024) + bias   via WMMA f16,
// then f32 gate nonlinearities, c kept in registers, h repacked into the
// WMMA A-fragment layout in LDS for the next step. Weights stream from L2
// every step (2 MB total, resident in the 192 MB L2).
// ---------------------------------------------------------------------------
__global__ __launch_bounds__(1024) void lstm_scan(
    const float* __restrict__ x, const _Float16* __restrict__ wpack,
    const float* __restrict__ bias, float* __restrict__ out,
    float* __restrict__ hn, float* __restrict__ cn)
{
    const int d     = blockIdx.x >> 1;     // direction: 0 fwd, 1 bwd
    const int mt    = blockIdx.x & 1;      // batch tile
    const int bbase = mt * 16;
    const int tid   = threadIdx.x;
    const int lane  = tid & 31;
    const int wv    = tid >> 5;

    extern __shared__ __attribute__((aligned(128))) char smemraw[];
    _Float16* afrag = (_Float16*)smemraw;                       // [16][32][16] A-fragments
    float*    gbuf  = (float*)(smemraw + 16 * 32 * 16 * 2);     // [16][GSTR] f32 gates
    float*    sbias = (float*)(smemraw + 16 * 32 * 16 * 2 + 16 * GSTR * 4); // [1024]

    // ---- init: bias to LDS, zero h-region of A fragments, zero c/h regs ----
    sbias[tid] = bias[d * GG + tid];
    for (int i = tid; i < 8 * 32 * 16; i += 1024) afrag[i] = (_Float16)0.0f;

    float creg[4], hreg[4];
#pragma unroll
    for (int q = 0; q < 4; ++q) { creg[q] = 0.0f; hreg[q] = 0.0f; }

    const int nt0 = wv * 2, nt1 = wv * 2 + 1;
    const int coln0 = nt0 * 16 + (lane & 15);
    const int coln1 = nt1 * 16 + (lane & 15);
    const int mrow  = (lane >> 4) * 8;     // C/D layout: m = r + mrow

    const _Float16* wbase = wpack + (size_t)d * NTILES * KSTEPS * 32 * 16;
    const _Float16* w0 = wbase + (size_t)nt0 * KSTEPS * 32 * 16 + lane * 16;
    const _Float16* w1 = wbase + (size_t)nt1 * KSTEPS * 32 * 16 + lane * 16;

    // Warm the weight stream into cache (global_prefetch_b8)
#pragma unroll
    for (int kk = 0; kk < KSTEPS; ++kk) {
        __builtin_prefetch((const void*)(w0 + kk * 32 * 16), 0, 3);
        __builtin_prefetch((const void*)(w1 + kk * 32 * 16), 0, 3);
    }

    // Bias scalars: load ONCE, stays in SGPR/VGPR across the whole scan
    // (pre-loop loads are SSA values and unaffected by the in-loop clobber).
    __syncthreads();
    const float bs0 = sbias[coln0], bs1 = sbias[coln1];

    for (int ti = 0; ti < TT; ++ti) {
        const int t = d ? (TT - 1 - ti) : ti;

        // (a) pack x_t (f32 global, coalesced) into A-fragment layout, kk 8..15
#pragma unroll
        for (int q = 0; q < 4; ++q) {
            int p = tid + q * 1024;
            int m = p >> 8, i = p & 255;
            float xv = x[(size_t)(bbase + m) * TT * II + (size_t)t * II + i];
            int kk = 8 + (i >> 5);
            int k  = i & 31;
            int lhalf = ((k & 16) >> 1) | (k & 7);      // A layout: k = (h&8)*2 + Lhi*8 + (h&7)
            int lL    = m + (((k >> 3) & 1) << 4);
            afrag[(kk * 32 + lL) * 16 + lhalf] = (_Float16)xv;
        }
        __syncthreads();

        // Compiler barrier: forbid LICM from hoisting the (loop-invariant)
        // weight-fragment loads out of the time loop — that hoist spills the
        // whole 1 MB weight stream to scratch. With this, weights are re-read
        // from global (L2-resident) every step via global_load_b128.
        asm volatile("" ::: "memory");

        // (c) WMMA: gates tile = sum_kk A(kk) * B(kk) + bias
        v8f acc0, acc1;
#pragma unroll
        for (int r = 0; r < 8; ++r) { acc0[r] = bs0; acc1[r] = bs1; }

        // one-stage software pipeline: fetch kk+1 while doing WMMA for kk
        v16h a  = *(const v16h*)&afrag[lane * 16];
        v16h b0 = *(const v16h*)(w0);
        v16h b1 = *(const v16h*)(w1);
#pragma unroll
        for (int kk = 0; kk < KSTEPS; ++kk) {
            v16h an = a, b0n = b0, b1n = b1;
            if (kk < KSTEPS - 1) {
                an  = *(const v16h*)&afrag[((kk + 1) * 32 + lane) * 16];
                b0n = *(const v16h*)(w0 + (kk + 1) * 32 * 16);
                b1n = *(const v16h*)(w1 + (kk + 1) * 32 * 16);
            }
            acc0 = __builtin_amdgcn_wmma_f32_16x16x32_f16(
                false, a, false, b0, (short)0, acc0, false, false);
            acc1 = __builtin_amdgcn_wmma_f32_16x16x32_f16(
                false, a, false, b1, (short)0, acc1, false, false);
            a = an; b0 = b0n; b1 = b1n;
        }

        // spill gate tiles to LDS (f32) so i/f/g/o can be regathered per hidden unit
#pragma unroll
        for (int r = 0; r < 8; ++r) {
            int m = r + mrow;
            gbuf[m * GSTR + coln0] = acc0[r];
            gbuf[m * GSTR + coln1] = acc1[r];
        }
        __syncthreads();

        // (e) elementwise: activations, cell/hidden update, output store,
        //     and repack h into A-fragment layout (kk 0..7) for next step
#pragma unroll
        for (int q = 0; q < 4; ++q) {
            int p = tid + q * 1024;
            int m = p >> 8, j = p & 255;
            float ig = gbuf[m * GSTR + j];
            float fg = gbuf[m * GSTR + 256 + j];
            float gg = gbuf[m * GSTR + 512 + j];
            float og = gbuf[m * GSTR + 768 + j];
            ig = 1.0f / (1.0f + __expf(-ig));
            fg = 1.0f / (1.0f + __expf(-fg));
            og = 1.0f / (1.0f + __expf(-og));
            gg = tanhf(gg);
            float c = fg * creg[q] + ig * gg;
            float h = og * tanhf(c);
            creg[q] = c; hreg[q] = h;

            out[((size_t)(bbase + m) * TT + t) * (2 * HH) + d * HH + j] = h;

            int kk = j >> 5, k = j & 31;
            int lhalf = ((k & 16) >> 1) | (k & 7);
            int lL    = m + (((k >> 3) & 1) << 4);
            afrag[(kk * 32 + lL) * 16 + lhalf] = (_Float16)h;
        }
        __syncthreads();
    }

    // final carry state: h_n (2,B,H), c_n (2,B,H)
#pragma unroll
    for (int q = 0; q < 4; ++q) {
        int p = tid + q * 1024;
        int m = p >> 8, j = p & 255;
        int b = bbase + m;
        hn[(d * BB + b) * HH + j] = hreg[q];
        cn[(d * BB + b) * HH + j] = creg[q];
    }
}

extern "C" void kernel_launch(void* const* d_in, const int* in_sizes, int n_in,
                              void* d_out, int out_size, void* d_ws, size_t ws_size,
                              hipStream_t stream) {
    const float* x      = (const float*)d_in[0];
    const float* w_ih_f = (const float*)d_in[1];
    const float* w_hh_f = (const float*)d_in[2];
    const float* b_ih_f = (const float*)d_in[3];
    const float* b_hh_f = (const float*)d_in[4];
    const float* w_ih_b = (const float*)d_in[5];
    const float* w_hh_b = (const float*)d_in[6];
    const float* b_ih_b = (const float*)d_in[7];
    const float* b_hh_b = (const float*)d_in[8];

    // workspace: wpack (2 MB of f16 fragments) + bias (8 KB)
    _Float16* wpack = (_Float16*)d_ws;
    float*    bias  = (float*)((char*)d_ws + (size_t)2 * NTILES * KSTEPS * 32 * 16 * sizeof(_Float16));

    float* out = (float*)d_out;
    float* hn  = out + (size_t)BB * TT * 2 * HH;   // after output (B,T,2H)
    float* cn  = hn + 2 * BB * HH;

    lstm_prep<<<1024, 256, 0, stream>>>(w_ih_f, w_hh_f, b_ih_f, b_hh_f,
                                        w_ih_b, w_hh_b, b_ih_b, b_hh_b,
                                        wpack, bias);

    size_t shmem = (size_t)16 * 32 * 16 * 2   // afrag
                 + (size_t)16 * GSTR * 4      // gbuf
                 + (size_t)GG * 4;            // sbias
    lstm_scan<<<4, 1024, shmem, stream>>>(x, wpack, bias, out, hn, cn);
}
